// ST_GCN_24919400251546
// MI455X (gfx1250) — matrix operand order repaired and checked by reading, not compiled, not run on hardware
//
#include <hip/hip_runtime.h>

// ---------------- problem constants ----------------
#define NN      6400          // nodes
#define TT      64            // frames
#define MM      409600        // NN*TT rows
#define NE      12288         // edges
#define NG      256           // graphs
#define NCLASS  60

typedef __attribute__((ext_vector_type(8)))  float          v8f;
typedef __attribute__((ext_vector_type(16))) __bf16         bf16x16;
typedef __attribute__((ext_vector_type(2)))  __bf16         bf16x2;
typedef __attribute__((ext_vector_type(16))) unsigned short ushort16;
typedef __attribute__((ext_vector_type(8)))  unsigned short ushort8;

union FragAB {
    ushort16 u;
    ushort8  u8[2];
    bf16x2   pk[8];
    bf16x16  b;
};

__device__ __forceinline__ unsigned short f32_to_bf16(float f) {
    unsigned int x = __float_as_uint(f);
    unsigned int r = x + 0x7FFFu + ((x >> 16) & 1u);   // round-to-nearest-even
    return (unsigned short)(r >> 16);
}

// ---------------- degree / normalization ----------------
__global__ void deg_kernel(const int* __restrict__ src, const int* __restrict__ dst,
                           float* __restrict__ dout, float* __restrict__ din) {
    int e = blockIdx.x * blockDim.x + threadIdx.x;
    if (e < NE) {
        atomicAdd(dout + src[e], 1.0f);
        atomicAdd(din  + dst[e], 1.0f);
    }
}

__global__ void invsqrt_kernel(float* __restrict__ a, float* __restrict__ b) {
    int i = blockIdx.x * blockDim.x + threadIdx.x;
    if (i < NN) {
        a[i] = rsqrtf(fmaxf(a[i], 1.0f));
        b[i] = rsqrtf(fmaxf(b[i], 1.0f));
    }
}

// ---------------- weight transpose+convert: W[K][F] f32 -> Wt[F][K] bf16 ----------------
__global__ void wtprep_kernel(const float* __restrict__ W, unsigned short* __restrict__ Wt,
                              int K, int F) {
    int idx = blockIdx.x * blockDim.x + threadIdx.x;
    if (idx < K * F) {
        int k = idx / F, n = idx - k * F;
        Wt[(size_t)n * K + k] = f32_to_bf16(W[idx]);
    }
}

// ---------------- edge scatter: agg[dst] += h[src] * inv_out[src] ----------------
// grid: (ceil(TF/4/256), NE), block 256; 4 floats per thread.
__global__ void scatter_kernel(const float* __restrict__ h, const int* __restrict__ src,
                               const int* __restrict__ dst, const float* __restrict__ inv_out,
                               float* __restrict__ agg, int TF) {
    int j = (blockIdx.x * blockDim.x + threadIdx.x) * 4;
    int e = blockIdx.y;
    if (j < TF) {
        int s = src[e], d = dst[e];
        float iv = inv_out[s];
        float4 v = *(const float4*)(h + (size_t)s * TF + j);
        float* ap = agg + (size_t)d * TF + j;
        atomicAdd(ap + 0, v.x * iv);
        atomicAdd(ap + 1, v.y * iv);
        atomicAdd(ap + 2, v.z * iv);
        atomicAdd(ap + 3, v.w * iv);
    }
}

// ---------------- layer 0: K=3 plain FMA GEMM, relu ----------------
__global__ void gemm3_kernel(const float* __restrict__ agg, const float* __restrict__ inv_in,
                             const float* __restrict__ W, const float* __restrict__ bias,
                             float* __restrict__ H) {
    int idx = blockIdx.x * blockDim.x + threadIdx.x;   // MM*64 total
    int m = idx >> 6, n = idx & 63;
    float inv = inv_in[m >> 6];
    const float* a = agg + (size_t)m * 3;
    float v = a[0] * inv * W[n] + a[1] * inv * W[64 + n] + a[2] * inv * W[128 + n] + bias[n];
    H[(size_t)m * 64 + n] = fmaxf(v, 0.0f);
}

// ---------------- WMMA GEMM: H = relu(scale(agg) @ W + b) [+ H] ----------------
// one wave == one 16-row M tile, all NT n-tiles.  K in {64,128,256}, Fout = NT*16.
template <int NT, bool RES>
__global__ void gemm_wmma_kernel(const float* __restrict__ A, const float* __restrict__ inv_in,
                                 const unsigned short* __restrict__ Wt,
                                 const float* __restrict__ bias, float* __restrict__ H,
                                 int K) {
    const int FOUT = NT * 16;
    int wave = blockIdx.x * (blockDim.x >> 5) + (threadIdx.x >> 5);
    int lane = threadIdx.x & 31;
    int lh   = lane >> 4;        // lane-half (K sub-band select)
    int r    = lane & 15;

    long m0 = (long)wave * 16;
    int  m  = (int)m0 + r;
    float inv = inv_in[m >> 6];            // node = row / 64
    const float* arow = A + (size_t)m * K;

    v8f acc[NT];
    v8f zero = {0.f, 0.f, 0.f, 0.f, 0.f, 0.f, 0.f, 0.f};
#pragma unroll
    for (int i = 0; i < NT; ++i) acc[i] = zero;

    for (int k0 = 0; k0 < K; k0 += 32) {
        int kb = k0 + lh * 8;
        // A fragment: rows striped per lane; lane-half picks K {0..7,16..23} / {8..15,24..31}
        float4 x0 = *(const float4*)(arow + kb);
        float4 x1 = *(const float4*)(arow + kb + 4);
        float4 x2 = *(const float4*)(arow + kb + 16);
        float4 x3 = *(const float4*)(arow + kb + 20);
        if (k0 + 32 < K) __builtin_prefetch(arow + kb + 32, 0, 3);

        float t[16] = {x0.x, x0.y, x0.z, x0.w, x1.x, x1.y, x1.z, x1.w,
                       x2.x, x2.y, x2.z, x2.w, x3.x, x3.y, x3.z, x3.w};
        FragAB a;
#if __has_builtin(__builtin_amdgcn_cvt_pk_bf16_f32)
#pragma unroll
        for (int i = 0; i < 8; ++i)
            a.pk[i] = __builtin_amdgcn_cvt_pk_bf16_f32(t[2 * i] * inv, t[2 * i + 1] * inv);
#else
#pragma unroll
        for (int i = 0; i < 16; ++i) a.u[i] = f32_to_bf16(t[i] * inv);
#endif

#pragma unroll
        for (int nt = 0; nt < NT; ++nt) {
            int n = nt * 16 + r;
            const unsigned short* brow = Wt + (size_t)n * K + k0 + lh * 8;
            FragAB b;
            b.u8[0] = *(const ushort8*)(brow);
            b.u8[1] = *(const ushort8*)(brow + 16);
            acc[nt] = __builtin_amdgcn_wmma_f32_16x16x32_bf16(
                false, a.b, false, b.b, (short)0, acc[nt], false, false);
        }
    }

    // epilogue: bias + relu (+ residual), C/D layout: M = rr + 8*lh, N = nt*16 + r
#pragma unroll
    for (int nt = 0; nt < NT; ++nt) {
        int col = nt * 16 + r;
        float bb = bias[col];
#pragma unroll
        for (int rr = 0; rr < 8; ++rr) {
            long mm = m0 + rr + 8 * lh;
            size_t idx = (size_t)mm * FOUT + col;
            float v = fmaxf(acc[nt][rr] + bb, 0.0f);
            if (RES) v += H[idx];
            H[idx] = v;
        }
    }
}

// ---------------- readout ----------------
__global__ void nodemean_kernel(const float* __restrict__ H, float* __restrict__ nm) {
    int idx = blockIdx.x * blockDim.x + threadIdx.x;   // NN*256 total
    int node = idx >> 8, f = idx & 255;
    const float* p = H + (size_t)node * TT * 256 + f;
    float s = 0.0f;
#pragma unroll 8
    for (int t = 0; t < TT; ++t) s += p[t * 256];
    nm[idx] = s * (1.0f / (float)TT);
}

__global__ void graphsum_kernel(const float* __restrict__ nm, const int* __restrict__ gid,
                                float* __restrict__ gsum) {
    int idx = blockIdx.x * blockDim.x + threadIdx.x;   // NN*256 total
    int node = idx >> 8, f = idx & 255;
    atomicAdd(gsum + (size_t)gid[node] * 256 + f, nm[idx]);
}

__global__ void graphcnt_kernel(const int* __restrict__ gid, float* __restrict__ cnt) {
    int i = blockIdx.x * blockDim.x + threadIdx.x;
    if (i < NN) atomicAdd(cnt + gid[i], 1.0f);
}

__global__ void head_kernel(const float* __restrict__ gsum, const float* __restrict__ cnt,
                            const float* __restrict__ Wfc, const float* __restrict__ bfc,
                            float* __restrict__ out) {
    __shared__ float ro[256];
    __shared__ float lg[64];
    int g = blockIdx.x, t = threadIdx.x;
    float ic = 1.0f / fmaxf(cnt[g], 1.0f);
    for (int f = t; f < 256; f += 64) ro[f] = gsum[(size_t)g * 256 + f] * ic;
    __syncthreads();
    float logit = -1e30f;
    if (t < NCLASS) {
        logit = bfc[t];
        for (int f = 0; f < 256; ++f) logit += ro[f] * Wfc[f * NCLASS + t];
    }
    lg[t] = logit;
    __syncthreads();
    float mx = -1e30f;
    for (int i = 0; i < NCLASS; ++i) mx = fmaxf(mx, lg[i]);
    float se = 0.0f;
    for (int i = 0; i < NCLASS; ++i) se += expf(lg[i] - mx);
    if (t < NCLASS) out[(size_t)g * NCLASS + t] = lg[t] - mx - logf(se);
}

// ---------------- host launch ----------------
static const int DIN[12]  = {3, 64, 64, 64, 64, 128, 128, 128, 128, 256, 256, 256};
static const int DOUT[12] = {64, 64, 64, 64, 128, 128, 128, 128, 256, 256, 256, 256};

extern "C" void kernel_launch(void* const* d_in, const int* in_sizes, int n_in,
                              void* d_out, int out_size, void* d_ws, size_t ws_size,
                              hipStream_t stream) {
    (void)in_sizes; (void)n_in; (void)out_size; (void)ws_size;

    const float* loc = (const float*)d_in[0];
    const int*   src = (const int*)d_in[1];
    const int*   dst = (const int*)d_in[2];
    const int*   gid = (const int*)d_in[3];
    const float* Wf[12]; const float* Bf[12];
    for (int i = 0; i < 12; ++i) { Wf[i] = (const float*)d_in[4 + 2 * i];
                                   Bf[i] = (const float*)d_in[5 + 2 * i]; }
    const float* Wfc = (const float*)d_in[28];
    const float* bfc = (const float*)d_in[29];
    float* out = (float*)d_out;

    // bump-allocate workspace (256B aligned)
    unsigned char* w = (unsigned char*)d_ws;
    size_t off = 0;
    auto alloc = [&](size_t bytes) { void* p = w + off; off = (off + bytes + 255) & ~(size_t)255; return p; };
    float* inv_out = (float*)alloc((size_t)NN * 4);
    float* inv_in  = (float*)alloc((size_t)NN * 4);
    float* agg     = (float*)alloc((size_t)MM * 256 * 4);
    float* h       = (float*)alloc((size_t)MM * 256 * 4);
    unsigned short* wt[12];
    for (int i = 1; i < 12; ++i) wt[i] = (unsigned short*)alloc((size_t)DIN[i] * DOUT[i] * 2);
    float* nm   = (float*)alloc((size_t)NN * 256 * 4);
    float* gsum = (float*)alloc((size_t)NG * 256 * 4);
    float* cnt  = (float*)alloc((size_t)NG * 4);

    // degrees -> inv sqrt
    hipMemsetAsync(inv_out, 0, (size_t)NN * 4, stream);
    hipMemsetAsync(inv_in,  0, (size_t)NN * 4, stream);
    deg_kernel<<<(NE + 255) / 256, 256, 0, stream>>>(src, dst, inv_out, inv_in);
    invsqrt_kernel<<<(NN + 255) / 256, 256, 0, stream>>>(inv_out, inv_in);

    // weight transpose+convert for WMMA layers
    for (int i = 1; i < 12; ++i) {
        int n = DIN[i] * DOUT[i];
        wtprep_kernel<<<(n + 255) / 256, 256, 0, stream>>>(Wf[i], wt[i], DIN[i], DOUT[i]);
    }

    // layer 0 (K=3, VALU path)
    hipMemsetAsync(agg, 0, (size_t)MM * 3 * 4, stream);
    scatter_kernel<<<dim3(1, NE), 256, 0, stream>>>(loc, src, dst, inv_out, agg, 192);
    gemm3_kernel<<<(MM * 64) / 256, 256, 0, stream>>>(agg, inv_in, Wf[0], Bf[0], h);

    // layers 1..11 (WMMA path, in-place h)
    const int blocks = (MM / 16) / 8;   // 8 waves/block, one 16-row tile per wave
    for (int i = 1; i < 12; ++i) {
        int K = DIN[i], F = DOUT[i], TF = TT * K;
        bool res = (i % 4) != 0;
        hipMemsetAsync(agg, 0, (size_t)MM * K * 4, stream);
        scatter_kernel<<<dim3((TF / 4 + 255) / 256, NE), 256, 0, stream>>>(h, src, dst, inv_out, agg, TF);
        switch (F / 16) {
            case 4:
                if (res) gemm_wmma_kernel<4, true ><<<blocks, 256, 0, stream>>>(agg, inv_in, wt[i], Bf[i], h, K);
                else     gemm_wmma_kernel<4, false><<<blocks, 256, 0, stream>>>(agg, inv_in, wt[i], Bf[i], h, K);
                break;
            case 8:
                if (res) gemm_wmma_kernel<8, true ><<<blocks, 256, 0, stream>>>(agg, inv_in, wt[i], Bf[i], h, K);
                else     gemm_wmma_kernel<8, false><<<blocks, 256, 0, stream>>>(agg, inv_in, wt[i], Bf[i], h, K);
                break;
            default:
                if (res) gemm_wmma_kernel<16, true ><<<blocks, 256, 0, stream>>>(agg, inv_in, wt[i], Bf[i], h, K);
                else     gemm_wmma_kernel<16, false><<<blocks, 256, 0, stream>>>(agg, inv_in, wt[i], Bf[i], h, K);
                break;
        }
    }

    // readout + head
    nodemean_kernel<<<(NN * 256) / 256, 256, 0, stream>>>(h, nm);
    hipMemsetAsync(gsum, 0, (size_t)NG * 256 * 4, stream);
    hipMemsetAsync(cnt,  0, (size_t)NG * 4, stream);
    graphsum_kernel<<<(NN * 256) / 256, 256, 0, stream>>>(nm, gid, gsum);
    graphcnt_kernel<<<(NN + 255) / 256, 256, 0, stream>>>(gid, cnt);
    head_kernel<<<NG, 64, 0, stream>>>(gsum, cnt, Wfc, bfc, out);
}
